// PointTokenizer_69544110457437
// MI455X (gfx1250) — matrix operand order repaired and checked by reading, not compiled, not run on hardware
//
#include <hip/hip_runtime.h>
#include <hip/hip_bf16.h>
#include <float.h>
#include <stdint.h>

typedef __attribute__((ext_vector_type(16))) __bf16 v16bf;
typedef __attribute__((ext_vector_type(8)))  float  v8f;

// Problem constants
#define NB   8
#define NPTS 8192
#define NM   512          // centers per batch
#define NK   32           // group size
#define NG   (NB*NM)      // 4096 groups
#define ROWS (NG*NK)      // 131072 MLP rows
#define C1   64
#define C2   128
#define C3   384

// ---------------------------------------------------------------------------
// Farthest point sampling: one block per batch, dist[] lives in LDS.
// ---------------------------------------------------------------------------
#define FPS_T 256
__global__ void fps_kernel(const float* __restrict__ pts, float* __restrict__ centers_out)
{
    __shared__ float dist[NPTS];
    __shared__ float red_v[FPS_T];
    __shared__ int   red_i[FPS_T];
    __shared__ float cen[3];

    const int b = blockIdx.x;
    const int t = threadIdx.x;
    const float* P = pts + (size_t)b * NPTS * 3;

    for (int i = t; i < NPTS; i += FPS_T) dist[i] = 1e10f;
    int far = 0;
    __syncthreads();

    for (int s = 0; s < NM; ++s) {
        if (t == 0) {
            float cx = P[far*3+0], cy = P[far*3+1], cz = P[far*3+2];
            cen[0] = cx; cen[1] = cy; cen[2] = cz;
            float* co = centers_out + ((size_t)b*NM + s) * 3;
            co[0] = cx; co[1] = cy; co[2] = cz;
        }
        __syncthreads();
        float cx = cen[0], cy = cen[1], cz = cen[2];
        float bv = -1.0f; int bi = 0;
        for (int i = t; i < NPTS; i += FPS_T) {
            float dx = P[i*3+0]-cx, dy = P[i*3+1]-cy, dz = P[i*3+2]-cz;
            float d  = dx*dx + dy*dy + dz*dz;
            float nd = fminf(dist[i], d);
            dist[i] = nd;
            if (nd > bv) { bv = nd; bi = i; }
        }
        red_v[t] = bv; red_i[t] = bi;
        __syncthreads();
        for (int off = FPS_T/2; off > 0; off >>= 1) {
            if (t < off) {
                float v2 = red_v[t+off]; int i2 = red_i[t+off];
                if (v2 > red_v[t] || (v2 == red_v[t] && i2 < red_i[t])) {
                    red_v[t] = v2; red_i[t] = i2;
                }
            }
            __syncthreads();
        }
        far = red_i[0];
        __syncthreads();
    }
}

// ---------------------------------------------------------------------------
// kNN grouping: one block per center; 32 argmin-select passes over LDS dists.
// ---------------------------------------------------------------------------
#define KNN_T 256
__global__ void knn_kernel(const float* __restrict__ pts, const float* __restrict__ centers,
                           float* __restrict__ grouped)
{
    __shared__ float d2[NPTS];
    __shared__ float red_v[KNN_T];
    __shared__ int   red_i[KNN_T];

    const int g = blockIdx.x;       // 0..NG-1
    const int b = g >> 9;           // g / NM
    const int t = threadIdx.x;
    const float* P = pts + (size_t)b * NPTS * 3;
    const float* C = centers + (size_t)g * 3;
    float cx = C[0], cy = C[1], cz = C[2];

    for (int i = t; i < NPTS; i += KNN_T) {
        float dx = P[i*3+0]-cx, dy = P[i*3+1]-cy, dz = P[i*3+2]-cz;
        d2[i] = dx*dx + dy*dy + dz*dz;
    }
    __syncthreads();

    for (int j = 0; j < NK; ++j) {
        float bv = 3.4e38f; int bi = 0;
        for (int i = t; i < NPTS; i += KNN_T) {
            float v = d2[i];
            if (v < bv) { bv = v; bi = i; }
        }
        red_v[t] = bv; red_i[t] = bi;
        __syncthreads();
        for (int off = KNN_T/2; off > 0; off >>= 1) {
            if (t < off) {
                float v2 = red_v[t+off]; int i2 = red_i[t+off];
                if (v2 < red_v[t] || (v2 == red_v[t] && i2 < red_i[t])) {
                    red_v[t] = v2; red_i[t] = i2;
                }
            }
            __syncthreads();
        }
        if (t == 0) {
            int sel = red_i[0];
            d2[sel] = 3.4e38f;
            float* o = grouped + ((size_t)g*NK + j) * 3;
            o[0] = P[sel*3+0]-cx; o[1] = P[sel*3+1]-cy; o[2] = P[sel*3+2]-cz;
        }
        __syncthreads();
    }
}

// ---------------------------------------------------------------------------
// Pack W2/W3 (f32, row-major [K,N]) into per-lane-contiguous WMMA B fragments.
// Fragment layout: idx = ((tile*KSTEPS + kstep)*32 + lane)*16 + j
//   value = W[(kstep*32 + (j>>3)*16 + (lane>>4)*8 + (j&7))*N + tile*16 + (lane&15)]
// ---------------------------------------------------------------------------
__global__ void pack_kernel(const float* __restrict__ W2, const float* __restrict__ W3,
                            __bf16* __restrict__ W2p, __bf16* __restrict__ W3p)
{
    int i = blockIdx.x*256 + threadIdx.x;
    if (i < C1*C2) {            // 8 tiles x 2 ksteps x 32 lanes x 16
        int j = i & 15, l = (i >> 4) & 31, ks = (i >> 9) & 1, tl = i >> 10;
        W2p[i] = (__bf16)W2[(size_t)(ks*32 + (j >> 3)*16 + (l >> 4)*8 + (j & 7))*C2 + tl*16 + (l & 15)];
    }
    if (i < C2*C3) {            // 24 tiles x 4 ksteps x 32 lanes x 16
        int j = i & 15, l = (i >> 4) & 31, ks = (i >> 9) & 3, tl = i >> 11;
        W3p[i] = (__bf16)W3[(size_t)(ks*32 + (j >> 3)*16 + (l >> 4)*8 + (j & 7))*C3 + tl*16 + (l & 15)];
    }
}

// ---------------------------------------------------------------------------
// Layer 1 stats only (K=3): per-channel sum/sumsq of h1 (h1 never stored)
// ---------------------------------------------------------------------------
__global__ void layer1_stats_kernel(const float* __restrict__ grouped, const float* __restrict__ W1,
                                    const float* __restrict__ b1,
                                    float* __restrict__ sum1, float* __restrict__ sq1)
{
    __shared__ float s_sum[C1], s_sq[C1];
    int t = threadIdx.x;
    if (t < C1) { s_sum[t] = 0.f; s_sq[t] = 0.f; }
    __syncthreads();

    size_t idx = (size_t)blockIdx.x*256 + t;   // < ROWS*C1
    int    c   = (int)(idx & (C1-1));
    size_t row = idx >> 6;
    const float* x = grouped + row*3;
    float v = b1[c] + x[0]*W1[c] + x[1]*W1[C1+c] + x[2]*W1[2*C1+c];
    atomicAdd(&s_sum[c], v);
    atomicAdd(&s_sq[c],  v*v);
    __syncthreads();
    if (t < C1) { atomicAdd(&sum1[t], s_sum[t]); atomicAdd(&sq1[t], s_sq[t]); }
}

// ---------------------------------------------------------------------------
// Fold BN1 into layer-1 weights: W1f = W1*sc, b1f = b1*sc + sh
// ---------------------------------------------------------------------------
__global__ void fold1_kernel(const float* __restrict__ sum, const float* __restrict__ sq,
                             const float* __restrict__ gamma, const float* __restrict__ beta,
                             const float* __restrict__ W1, const float* __restrict__ b1,
                             float n, float* __restrict__ W1f, float* __restrict__ b1f)
{
    int c = threadIdx.x;                // 64 threads
    float m  = sum[c]/n;
    float v  = sq[c]/n - m*m;
    float sc = gamma[c]*rsqrtf(v + 1e-5f);
    float sh = beta[c] - m*sc;
    W1f[c]        = W1[c]*sc;
    W1f[C1+c]     = W1[C1+c]*sc;
    W1f[2*C1+c]   = W1[2*C1+c]*sc;
    b1f[c]        = b1[c]*sc + sh;
}

// ---------------------------------------------------------------------------
// BN fold (layers 2,3): scale = gamma*rsqrt(var+eps), shift = beta - mean*scale
// ---------------------------------------------------------------------------
__global__ void stats_kernel(const float* __restrict__ sum, const float* __restrict__ sq,
                             const float* __restrict__ gamma, const float* __restrict__ beta,
                             float n, float* __restrict__ scale, float* __restrict__ shift, int C)
{
    int c = blockIdx.x*blockDim.x + threadIdx.x;
    if (c >= C) return;
    float m  = sum[c]/n;
    float v  = sq[c]/n - m*m;
    float sc = gamma[c]*rsqrtf(v + 1e-5f);
    scale[c] = sc;
    shift[c] = beta[c] - m*sc;
}

// ---------------------------------------------------------------------------
// Init per-(group,channel) running max/min
// ---------------------------------------------------------------------------
__global__ void init_minmax_kernel(float* __restrict__ gmax, float* __restrict__ gmin)
{
    size_t idx = (size_t)blockIdx.x*256 + threadIdx.x;   // < NG*C3
    gmax[idx] = -FLT_MAX;
    gmin[idx] =  FLT_MAX;
}

// ---------------------------------------------------------------------------
// GEMM2: h2 = relu(bn1(grouped@W1+b1)) @ W2 + b2   (layer-1 recomputed on the
// fly with folded weights).  v_wmma_f32_16x16x32_bf16, K=64, 8 waves/block.
// ---------------------------------------------------------------------------
__global__ void gemm2_kernel(const float* __restrict__ grouped,
                             const float* __restrict__ W1f, const float* __restrict__ b1f,
                             const __bf16* __restrict__ Wb, const float* __restrict__ bias,
                             float* __restrict__ h2, float* __restrict__ sum2, float* __restrict__ sq2)
{
    __shared__ float s_sum[C2], s_sq[C2];
    __shared__ float s_w1[3*C1], s_b1[C1];
    int t = threadIdx.x;
    if (t < C2) { s_sum[t] = 0.f; s_sq[t] = 0.f; }
    if (t < 3*C1) s_w1[t] = W1f[t];
    if (t < C1)   s_b1[t] = b1f[t];
    __syncthreads();

    int wave = t >> 5;           // 0..7 -> column tile
    int lane = t & 31;
    int lrow = lane & 15;
    int half = lane >> 4;
    size_t row0 = (size_t)blockIdx.x * 16;
    int col = wave*16 + lrow;

    const float* xr = grouped + (row0 + lrow)*3;
    float x0 = xr[0], x1 = xr[1], x2 = xr[2];

    v8f acc;
    float bv = bias[col];
    #pragma unroll
    for (int r = 0; r < 8; ++r) acc[r] = bv;

    #pragma unroll
    for (int kk = 0; kk < C1; kk += 32) {
        v16bf a;
        #pragma unroll
        for (int j = 0; j < 8; ++j) {
            int k0 = kk + half*8 + j;
            float h0 = s_b1[k0] + x0*s_w1[k0] + x1*s_w1[C1+k0] + x2*s_w1[2*C1+k0];
            a[j] = (__bf16)fmaxf(h0, 0.f);
            int k1 = kk + 16 + half*8 + j;
            float h1 = s_b1[k1] + x0*s_w1[k1] + x1*s_w1[C1+k1] + x2*s_w1[2*C1+k1];
            a[8+j] = (__bf16)fmaxf(h1, 0.f);
        }
        v16bf bmat = *(const v16bf*)(Wb + ((size_t)(wave*2 + (kk >> 5))*32 + lane)*16);
        acc = __builtin_amdgcn_wmma_f32_16x16x32_bf16(false, a, false, bmat,
                                                      (short)0, acc, false, false);
    }

    float ls = 0.f, lq = 0.f;
    #pragma unroll
    for (int r = 0; r < 8; ++r) {
        float v = acc[r];
        int m = half*8 + r;
        h2[(row0 + m)*C2 + col] = v;
        ls += v; lq += v*v;
    }
    atomicAdd(&s_sum[col], ls);
    atomicAdd(&s_sq[col],  lq);
    __syncthreads();
    if (t < C2) { atomicAdd(&sum2[t], s_sum[t]); atomicAdd(&sq2[t], s_sq[t]); }
}

// ---------------------------------------------------------------------------
// GEMM3: raw h3 tile = relu(bn2(h2)) @ W3 + b3, K=128, N=384 (3 tiles/wave).
// The 16x128 f32 A tile (contiguous 8KB of h2) is staged into LDS ONCE per
// block via async global->LDS copies (ASYNCcnt / s_wait_asynccnt), instead of
// 8 waves each re-reading it from global.  No h3 store: emits per-(group,
// channel) running max/min + BN3 stats.  All 16 rows of a block are one group.
// ---------------------------------------------------------------------------
__global__ void gemm3_kernel(const float* __restrict__ h2, const __bf16* __restrict__ Wb,
                             const float* __restrict__ scale, const float* __restrict__ shift,
                             const float* __restrict__ bias,
                             float* __restrict__ gmax, float* __restrict__ gmin,
                             float* __restrict__ sum3, float* __restrict__ sq3)
{
    __shared__ float s_a[16*C2];            // 8KB raw A tile, row-major
    __shared__ float s_sum[C3], s_sq[C3];
    int t = threadIdx.x;
    for (int c = t; c < C3; c += 256) { s_sum[c] = 0.f; s_sq[c] = 0.f; }

    size_t row0 = (size_t)blockIdx.x * 16;
    size_t grp  = row0 >> 5;                // group id (constant per block)

    // ---- async stage: 256 threads x 32B = 8KB (2 x B128 each) ----
    {
        int e = t * 8;                                   // flat f32 index in tile
        const float* src = h2 + row0*C2 + e;             // tile is contiguous
        unsigned      ldst = (unsigned)(uintptr_t)(s_a + e);   // low 32b = LDS offset
        unsigned long long ga = (unsigned long long)(uintptr_t)src;
        asm volatile("global_load_async_to_lds_b128 %0, %1, off"
                     :: "v"(ldst), "v"(ga) : "memory");
        asm volatile("global_load_async_to_lds_b128 %0, %1, off offset:16"
                     :: "v"(ldst), "v"(ga) : "memory");
        asm volatile("s_wait_asynccnt 0x0" ::: "memory");
    }
    __syncthreads();

    int wave = t >> 5;
    int lane = t & 31;
    int lrow = lane & 15;
    int half = lane >> 4;

    v8f acc[3];
    int cols[3];
    #pragma unroll
    for (int u = 0; u < 3; ++u) {
        cols[u] = (wave*3 + u)*16 + lrow;
        float bv = bias[cols[u]];
        #pragma unroll
        for (int r = 0; r < 8; ++r) acc[u][r] = bv;
    }

    for (int kk = 0; kk < C2; kk += 32) {
        v16bf a;
        const float* arow = s_a + lrow*C2 + kk;          // ds_load from LDS
        #pragma unroll
        for (int j = 0; j < 8; ++j) {
            int k0 = half*8 + j;
            a[j]   = (__bf16)fmaxf(arow[k0]*scale[kk+k0] + shift[kk+k0], 0.f);
            int k1 = 16 + half*8 + j;
            a[8+j] = (__bf16)fmaxf(arow[k1]*scale[kk+k1] + shift[kk+k1], 0.f);
        }
        #pragma unroll
        for (int u = 0; u < 3; ++u) {
            v16bf bmat = *(const v16bf*)(Wb +
                ((size_t)((wave*3 + u)*4 + (kk >> 5))*32 + lane)*16);
            acc[u] = __builtin_amdgcn_wmma_f32_16x16x32_bf16(false, a, false, bmat,
                                                             (short)0, acc[u], false, false);
        }
    }

    #pragma unroll
    for (int u = 0; u < 3; ++u) {
        float ls = 0.f, lq = 0.f;
        float mx = -FLT_MAX, mn = FLT_MAX;
        #pragma unroll
        for (int r = 0; r < 8; ++r) {
            float v = acc[u][r];
            ls += v; lq += v*v;
            mx = fmaxf(mx, v); mn = fminf(mn, v);
        }
        atomicAdd(&s_sum[cols[u]], ls);
        atomicAdd(&s_sq[cols[u]],  lq);
        atomicMax(&gmax[grp*C3 + cols[u]], mx);
        atomicMin(&gmin[grp*C3 + cols[u]], mn);
    }
    __syncthreads();
    for (int c = t; c < C3; c += 256) { atomicAdd(&sum3[c], s_sum[c]); atomicAdd(&sq3[c], s_sq[c]); }
}

// ---------------------------------------------------------------------------
// tokens = affine(BN3) of running max/min (monotone map)
// ---------------------------------------------------------------------------
__global__ void token_kernel(const float* __restrict__ gmax, const float* __restrict__ gmin,
                             const float* __restrict__ scale, const float* __restrict__ shift,
                             float* __restrict__ tokens)
{
    size_t idx = (size_t)blockIdx.x*256 + threadIdx.x;   // < NG*C3
    int c = (int)(idx % C3);
    float sc = scale[c], sh = shift[c];
    tokens[idx] = (sc >= 0.f ? gmax[idx] : gmin[idx]) * sc + sh;
}

// ---------------------------------------------------------------------------
// Host orchestration
// ---------------------------------------------------------------------------
extern "C" void kernel_launch(void* const* d_in, const int* in_sizes, int n_in,
                              void* d_out, int out_size, void* d_ws, size_t ws_size,
                              hipStream_t stream)
{
    const float* points = (const float*)d_in[0];
    const float* W1 = (const float*)d_in[1];
    const float* b1 = (const float*)d_in[2];
    const float* g1 = (const float*)d_in[3];
    const float* be1= (const float*)d_in[4];
    const float* W2 = (const float*)d_in[5];
    const float* b2 = (const float*)d_in[6];
    const float* g2 = (const float*)d_in[7];
    const float* be2= (const float*)d_in[8];
    const float* W3 = (const float*)d_in[9];
    const float* b3 = (const float*)d_in[10];
    const float* g3 = (const float*)d_in[11];
    const float* be3= (const float*)d_in[12];

    float* tokens  = (float*)d_out;                       // [NG, C3]
    float* centers = (float*)d_out + (size_t)NG*C3;       // [NG, 3]

    char* ws = (char*)d_ws;
    size_t o = 0;
    float*  grouped = (float*)(ws + o); o += (size_t)NG*NK*3*4;     // 1.5 MB
    float*  h2      = (float*)(ws + o); o += (size_t)ROWS*C2*4;     // 64 MB
    float*  gmax    = (float*)(ws + o); o += (size_t)NG*C3*4;       // 6.3 MB
    float*  gmin    = (float*)(ws + o); o += (size_t)NG*C3*4;       // 6.3 MB
    __bf16* W2p     = (__bf16*)(ws + o); o += (size_t)C1*C2*2;
    __bf16* W3p     = (__bf16*)(ws + o); o += (size_t)C2*C3*2;
    float*  W1f     = (float*)(ws + o); o += 3*C1*4;
    float*  b1f     = (float*)(ws + o); o += C1*4;
    float*  stats   = (float*)(ws + o);
    float*  sum1 = stats;        float* sq1 = sum1 + C1;
    float*  sum2 = sq1 + C1;     float* sq2 = sum2 + C2;
    float*  sum3 = sq2 + C2;     float* sq3 = sum3 + C3;
    o += (size_t)(2*(C1+C2+C3))*4;
    float*  aff     = (float*)(ws + o);
    float*  scale2 = aff;            float* shift2 = scale2 + C2;
    float*  scale3 = shift2 + C2;    float* shift3 = scale3 + C3;

    const float n = (float)ROWS;

    fps_kernel<<<NB, FPS_T, 0, stream>>>(points, centers);
    knn_kernel<<<NG, KNN_T, 0, stream>>>(points, centers, grouped);
    pack_kernel<<<(C2*C3 + 255)/256, 256, 0, stream>>>(W2, W3, W2p, W3p);
    hipMemsetAsync(stats, 0, (size_t)(2*(C1+C2+C3))*4, stream);
    init_minmax_kernel<<<(int)((size_t)NG*C3/256), 256, 0, stream>>>(gmax, gmin);

    layer1_stats_kernel<<<(int)((size_t)ROWS*C1/256), 256, 0, stream>>>(grouped, W1, b1, sum1, sq1);
    fold1_kernel<<<1, C1, 0, stream>>>(sum1, sq1, g1, be1, W1, b1, n, W1f, b1f);

    gemm2_kernel<<<ROWS/16, 256, 0, stream>>>(grouped, W1f, b1f, W2p, b2, h2, sum2, sq2);
    stats_kernel<<<1, C2, 0, stream>>>(sum2, sq2, g2, be2, n, scale2, shift2, C2);

    gemm3_kernel<<<ROWS/16, 256, 0, stream>>>(h2, W3p, scale2, shift2, b3,
                                              gmax, gmin, sum3, sq3);
    stats_kernel<<<(C3 + 255)/256, 256, 0, stream>>>(sum3, sq3, g3, be3, n, scale3, shift3, C3);

    token_kernel<<<(int)((size_t)NG*C3/256), 256, 0, stream>>>(gmax, gmin, scale3, shift3, tokens);
}
